// _UserModel_11613591569171
// MI455X (gfx1250) — matrix operand order repaired and checked by reading, not compiled, not run on hardware
//
#include <hip/hip_runtime.h>
#include <hip/hip_bf16.h>

// ---------------------------------------------------------------------------
// CDNA5 (gfx1250) WMMA implementation of the 7-branch social-attention model.
// v_wmma_f32_16x16x32_f16 with f32 accumulation. Weights are pre-transposed
// to f16 once (prep kernel) so every B fragment is two 16B loads; A fragments
// use 128-bit loads; LDS staging (fenced with s_wait_dscnt) re-layouts D->A
// between chained GEMMs; wave32 shuffle reductions do the masked
// exp-normalize attention.
// ---------------------------------------------------------------------------

typedef __attribute__((ext_vector_type(16))) _Float16 v16h;
typedef __attribute__((ext_vector_type(8)))  _Float16 v8h;
typedef __attribute__((ext_vector_type(8)))  float    v8f;

#define DEMB 64

__device__ __forceinline__ void lds_wave_fence() {
#if defined(__gfx1250__)
  asm volatile("s_wait_dscnt 0" ::: "memory");
#else
  __threadfence_block();
#endif
}

__device__ __forceinline__ v8f wmma_f16(v16h a, v16h b, v8f c) {
#if defined(__gfx1250__)
  return __builtin_amdgcn_wmma_f32_16x16x32_f16(false, a, false, b, (short)0, c,
                                                false, false);
#else
  return c;
#endif
}

__device__ __forceinline__ v16h cat8(v8h lo, v8h hi) {
  return __builtin_shufflevector(lo, hi, 0, 1, 2, 3, 4, 5, 6, 7, 8, 9, 10, 11,
                                 12, 13, 14, 15);
}

__device__ __forceinline__ v16h a_frag_zero() {
  v16h z;
#pragma unroll
  for (int i = 0; i < 16; ++i) z[i] = (_Float16)0.0f;
  return z;
}

// A fragment (16x32 f16) for K-chunk kc from a per-lane f32 row pointer.
// Lane L supplies row m=L%16; its 16 elements are two runs of 8 consecutive k:
//   run0 at k = 32kc + 8*half, run1 at k = 32kc + 16 + 8*half   (half = L/16)
__device__ __forceinline__ v16h a_frag_row(const float* rowp, int kc, int lane) {
  const int half = (lane >> 4) & 1;
  const float* p0 = rowp + kc * 32 + half * 8;
  const float4 x0 = *(const float4*)p0;
  const float4 x1 = *(const float4*)(p0 + 4);
  const float4 x2 = *(const float4*)(p0 + 16);
  const float4 x3 = *(const float4*)(p0 + 20);
  v16h a;
  a[0] = (_Float16)x0.x;  a[1] = (_Float16)x0.y;
  a[2] = (_Float16)x0.z;  a[3] = (_Float16)x0.w;
  a[4] = (_Float16)x1.x;  a[5] = (_Float16)x1.y;
  a[6] = (_Float16)x1.z;  a[7] = (_Float16)x1.w;
  a[8] = (_Float16)x2.x;  a[9] = (_Float16)x2.y;
  a[10] = (_Float16)x2.z; a[11] = (_Float16)x2.w;
  a[12] = (_Float16)x3.x; a[13] = (_Float16)x3.y;
  a[14] = (_Float16)x3.z; a[15] = (_Float16)x3.w;
  return a;
}

// A fragment from the f16 LDS staging tile: two ds_load_b128 per lane.
__device__ __forceinline__ v16h a_frag_stage(const _Float16* st, int ldc, int kc,
                                             int lane) {
  const int m = lane & 15, half = (lane >> 4) & 1;
  const _Float16* row = st + m * ldc + kc * 32 + half * 8;
  return cat8(*(const v8h*)row, *(const v8h*)(row + 16));
}

// B fragment (32x16 f16) from a pre-transposed f16 weight Wt[dout x din]
// (Wt[n*din + k] = W[k*dout + n]): lane L holds column n = 16nt + L%16,
// K-runs k = koff + 32kc + 8*(L/16) + j and +16 -> two 16B loads.
__device__ __forceinline__ v16h b_frag_t(const _Float16* Wt, int din, int koff,
                                         int kc, int nt, int lane) {
  const int n = nt * 16 + (lane & 15);
  const int kb = koff + kc * 32 + ((lane >> 4) & 1) * 8;
  const _Float16* p = Wt + (size_t)n * din + kb;
  return cat8(*(const v8h*)p, *(const v8h*)(p + 16));
}

// C/D layout: (vgpr r, lane L) -> element (m = r + 8*(L/16), n = 16nt + L%16)
__device__ __forceinline__ v8f acc_bias(const float* bias, int nt, int lane) {
  const float bv = bias[nt * 16 + (lane & 15)];
  v8f a;
#pragma unroll
  for (int r = 0; r < 8; ++r) a[r] = bv;
  return a;
}

// Store accumulators to row-major f16 staging. act: 0 none, 1 leaky(0.2), 2 relu
__device__ __forceinline__ void stage_acc(_Float16* st, int ldc, const v8f* acc,
                                          int nts, int lane, int act) {
  const int nl = lane & 15, half = (lane >> 4) & 1;
  for (int nt = 0; nt < nts; ++nt) {
#pragma unroll
    for (int r = 0; r < 8; ++r) {
      float v = acc[nt][r];
      if (act == 1) v = v > 0.0f ? v : 0.2f * v;
      if (act == 2) v = v > 0.0f ? v : 0.0f;
      st[(r + 8 * half) * ldc + nt * 16 + nl] = (_Float16)v;
    }
  }
}

// ---------------------------------------------------------------------------
// Weight prep: W[din x dout] f32 (row-major) -> Wt[dout x din] f16.
// ---------------------------------------------------------------------------
struct PrepEntry { const float* src; unsigned dstoff; int din; int dout; };
struct PrepArgs  { PrepEntry e[25]; int n; };

__global__ __launch_bounds__(256) void prep_weights(PrepArgs a, _Float16* dst) {
  for (int i = 0; i < a.n; ++i) {
    const int din = a.e[i].din, dout = a.e[i].dout;
    const int total = din * dout;
    const float* src = a.e[i].src;
    _Float16* d = dst + a.e[i].dstoff;
    for (int idx = blockIdx.x * blockDim.x + threadIdx.x; idx < total;
         idx += gridDim.x * blockDim.x) {
      const int n = idx / din, k = idx - n * din;
      d[idx] = (_Float16)src[(size_t)k * dout + n];
    }
  }
}

// ---------------------------------------------------------------------------
// Generic attention branch.
// ---------------------------------------------------------------------------
struct BranchCfg {
  const float* ue; const float* ie; const float* re;
  const int* pair_ids;                          // in_mode 0: (item,rate) pairs
  const float* in_rows;                         // in_mode 1: f32 rows
  const int* in_row_ids; int in_row_id_stride;  // in_mode 2: gather ue rows
  int in_mode;
  const int* mask_ids; int mask_stride;         // mask = ids[..] > 0
  const int* parent_ids; int parent_stride;     // rank-1 parent (or per-row)
  int second_per_row;
  const _Float16 *Wg1t, *Wg2t;  const float *bg1, *bg2;  // g_v (in_mode 0)
  const _Float16 *Wkt;          const float *bk;         // w1..w7
  const _Float16 *Wa1t;         const float *ba1;        // att l1 (din=128)
  const float *Wa2, *ba2;                                // att l2 (64->1)
  const _Float16 *Waggt;        const float *bagg;       // aggregation
  float* out; int out_stride; int out_off;
  int groups; int rows;
};

__global__ __launch_bounds__(256) void branch_kernel(BranchCfg c) {
  __shared__ __attribute__((aligned(16))) _Float16 s_stage[8][16 * DEMB];
  __shared__ float s_vec[8][256];               // pW | t0 | t1 | v
  const int lane = threadIdx.x & 31;
  const int wave = threadIdx.x >> 5;
  _Float16* stg = s_stage[wave];
  float* sv = s_vec[wave];
  const int nl = lane & 15;
  const int half = (lane >> 4) & 1;
  const int wave0 = blockIdx.x * 8 + wave;
  const int nwaves = gridDim.x * 8;

  for (int g = wave0; g < c.groups; g += nwaves) {
    // ---- rank-1 second operand: wp = mask*(parent@Wk) + bk
    //      => att contribution is t0[n] + mask*t1[n]
    if (!c.second_per_row) {
      const int pidx = c.parent_ids[(size_t)g * c.parent_stride];
      const float* pv = c.ue + (size_t)pidx * DEMB;
      const int n0 = lane * 2;
      const _Float16* w0 = c.Wkt + (size_t)n0 * DEMB;
      const _Float16* w1 = w0 + DEMB;
      float a0 = 0.f, a1 = 0.f;
      for (int k = 0; k < DEMB; ++k) {
        const float p = pv[k];
        a0 += p * (float)w0[k];
        a1 += p * (float)w1[k];
      }
      sv[n0] = a0; sv[n0 + 1] = a1;
      lds_wave_fence();
      const _Float16* wb0 = c.Wa1t + (size_t)n0 * 128 + DEMB;  // bottom half
      const _Float16* wb1 = wb0 + 128;
      float t0a = c.ba1[n0], t0b = c.ba1[n0 + 1], t1a = 0.f, t1b = 0.f;
      for (int k = 0; k < DEMB; ++k) {
        const float pw = sv[k], bkk = c.bk[k];
        t1a += pw * (float)wb0[k]; t1b += pw * (float)wb1[k];
        t0a += bkk * (float)wb0[k]; t0b += bkk * (float)wb1[k];
      }
      sv[64 + n0]  = t0a; sv[64 + n0 + 1]  = t0b;
      sv[128 + n0] = t1a; sv[128 + n0 + 1] = t1b;
      lds_wave_fence();
    }

    float wsum[4] = {0.f, 0.f, 0.f, 0.f};
    float denom_part = 0.f;
    const float wa2v[4] = {c.Wa2[nl], c.Wa2[16 + nl], c.Wa2[32 + nl], c.Wa2[48 + nl]};
    const float ba2v = c.ba2[0];
    const int ntiles = (c.rows + 15) >> 4;

    for (int t = 0; t < ntiles; ++t) {
      const int mrow = t * 16 + nl;           // row this lane supplies to A
      const bool valid = mrow < c.rows;

      // ---- A fragments of the (possibly g_v-transformed) input rows
      v16h ain0, ain1;
      if (c.in_mode == 0) {
        v16h ax[4];
        if (valid) {
          const size_t pb = ((size_t)g * c.rows + mrow) * 2;
          const int iid = c.pair_ids[pb];
          const int rid = c.pair_ids[pb + 1];
          const float* ri = c.ie + (size_t)iid * DEMB;
          const float* rr = c.re + (size_t)rid * DEMB - DEMB;  // k in [64,128)
          ax[0] = a_frag_row(ri, 0, lane);
          ax[1] = a_frag_row(ri, 1, lane);
          ax[2] = a_frag_row(rr, 2, lane);
          ax[3] = a_frag_row(rr, 3, lane);
        } else {
          ax[0] = ax[1] = ax[2] = ax[3] = a_frag_zero();
        }
        // g_v.l1 : [16x128]x[128x64]+b, leaky(0.2)
        v8f t4[4];
        for (int nt = 0; nt < 4; ++nt) {
          v8f acc = acc_bias(c.bg1, nt, lane);
          for (int kc = 0; kc < 4; ++kc)
            acc = wmma_f16(ax[kc], b_frag_t(c.Wg1t, 128, 0, kc, nt, lane), acc);
          t4[nt] = acc;
        }
        stage_acc(stg, DEMB, t4, 4, lane, 1);
        lds_wave_fence();
        const v16h ah0 = a_frag_stage(stg, DEMB, 0, lane);
        const v16h ah1 = a_frag_stage(stg, DEMB, 1, lane);
        // g_v.l2 : [16x64]x[64x64]+b
        for (int nt = 0; nt < 4; ++nt) {
          v8f acc = acc_bias(c.bg2, nt, lane);
          acc = wmma_f16(ah0, b_frag_t(c.Wg2t, DEMB, 0, 0, nt, lane), acc);
          acc = wmma_f16(ah1, b_frag_t(c.Wg2t, DEMB, 0, 1, nt, lane), acc);
          t4[nt] = acc;
        }
        lds_wave_fence();
        stage_acc(stg, DEMB, t4, 4, lane, 0);
        lds_wave_fence();
        ain0 = a_frag_stage(stg, DEMB, 0, lane);
        ain1 = a_frag_stage(stg, DEMB, 1, lane);
      } else if (c.in_mode == 1) {
        const float* rowp = c.in_rows + ((size_t)g * c.rows + mrow) * DEMB;
        ain0 = valid ? a_frag_row(rowp, 0, lane) : a_frag_zero();
        ain1 = valid ? a_frag_row(rowp, 1, lane) : a_frag_zero();
      } else {
        const int rid =
            valid ? c.in_row_ids[((size_t)g * c.rows + mrow) * c.in_row_id_stride] : 0;
        const float* rowp = c.ue + (size_t)rid * DEMB;
        ain0 = valid ? a_frag_row(rowp, 0, lane) : a_frag_zero();
        ain1 = valid ? a_frag_row(rowp, 1, lane) : a_frag_zero();
      }

      // ---- wx = x @ Wk + bk  (kept in registers for the weighted sum)
      v8f wx[4];
      for (int nt = 0; nt < 4; ++nt) {
        v8f acc = acc_bias(c.bk, nt, lane);
        acc = wmma_f16(ain0, b_frag_t(c.Wkt, DEMB, 0, 0, nt, lane), acc);
        acc = wmma_f16(ain1, b_frag_t(c.Wkt, DEMB, 0, 1, nt, lane), acc);
        wx[nt] = acc;
      }
      lds_wave_fence();
      stage_acc(stg, DEMB, wx, 4, lane, 0);
      lds_wave_fence();
      const v16h awx0 = a_frag_stage(stg, DEMB, 0, lane);
      const v16h awx1 = a_frag_stage(stg, DEMB, 1, lane);

      // ---- per-row masks for the 8 rows this lane owns in C layout
      float mask8[8];
#pragma unroll
      for (int r = 0; r < 8; ++r) {
        const int m = t * 16 + r + 8 * half;
        const int id =
            (m < c.rows) ? c.mask_ids[((size_t)g * c.rows + m) * c.mask_stride] : 0;
        mask8[r] = id > 0 ? 1.0f : 0.0f;
      }

      // ---- attention l1 pre-activation
      v8f att[4];
      if (!c.second_per_row) {
        for (int nt = 0; nt < 4; ++nt) {
          const float t0 = sv[64 + nt * 16 + nl];
          const float t1 = sv[128 + nt * 16 + nl];
          v8f acc;
#pragma unroll
          for (int r = 0; r < 8; ++r) acc[r] = t0 + mask8[r] * t1;
          att[nt] = acc;
        }
      } else {
        for (int nt = 0; nt < 4; ++nt) att[nt] = acc_bias(c.ba1, nt, lane);
      }
      for (int nt = 0; nt < 4; ++nt) {
        att[nt] = wmma_f16(awx0, b_frag_t(c.Wa1t, 128, 0, 0, nt, lane), att[nt]);
        att[nt] = wmma_f16(awx1, b_frag_t(c.Wa1t, 128, 0, 1, nt, lane), att[nt]);
      }
      if (c.second_per_row) {  // branch 5: wp = ue[pid_row] @ Wk + bk (no mask)
        const int pid =
            valid ? c.parent_ids[((size_t)g * c.rows + mrow) * c.parent_stride] : 0;
        const float* prow = c.ue + (size_t)pid * DEMB;
        const v16h ap0 = valid ? a_frag_row(prow, 0, lane) : a_frag_zero();
        const v16h ap1 = valid ? a_frag_row(prow, 1, lane) : a_frag_zero();
        v8f wp[4];
        for (int nt = 0; nt < 4; ++nt) {
          v8f acc = acc_bias(c.bk, nt, lane);
          acc = wmma_f16(ap0, b_frag_t(c.Wkt, DEMB, 0, 0, nt, lane), acc);
          acc = wmma_f16(ap1, b_frag_t(c.Wkt, DEMB, 0, 1, nt, lane), acc);
          wp[nt] = acc;
        }
        lds_wave_fence();
        stage_acc(stg, DEMB, wp, 4, lane, 0);
        lds_wave_fence();
        const v16h awp0 = a_frag_stage(stg, DEMB, 0, lane);
        const v16h awp1 = a_frag_stage(stg, DEMB, 1, lane);
        for (int nt = 0; nt < 4; ++nt) {  // bottom half of Wa1 (k offset 64)
          att[nt] = wmma_f16(awp0, b_frag_t(c.Wa1t, 128, 64, 0, nt, lane), att[nt]);
          att[nt] = wmma_f16(awp1, b_frag_t(c.Wa1t, 128, 64, 1, nt, lane), att[nt]);
        }
      }

      // ---- leaky -> att.l2 (64->1) -> masked exp; accumulate weighted sums
#pragma unroll
      for (int r = 0; r < 8; ++r) {
        float part = 0.f;
#pragma unroll
        for (int nt = 0; nt < 4; ++nt) {
          float v = att[nt][r];
          v = v > 0.f ? v : 0.2f * v;
          part += v * wa2v[nt];
        }
        part += __shfl_xor(part, 1);
        part += __shfl_xor(part, 2);
        part += __shfl_xor(part, 4);
        part += __shfl_xor(part, 8);
        const float s = __expf(part + ba2v) * mask8[r];
        denom_part += s;
#pragma unroll
        for (int nt = 0; nt < 4; ++nt) wsum[nt] += s * wx[nt][r];
      }
    }

    // ---- normalize, aggregation linear + relu, write out
    const float denom = denom_part + __shfl_xor(denom_part, 16) + 1e-10f;
#pragma unroll
    for (int nt = 0; nt < 4; ++nt) wsum[nt] += __shfl_xor(wsum[nt], 16);
    const float inv = 1.0f / denom;
    if (lane < 16) {
#pragma unroll
      for (int nt = 0; nt < 4; ++nt) sv[192 + nt * 16 + nl] = wsum[nt] * inv;
    }
    lds_wave_fence();
    const int j = lane * 2;
    const _Float16* wj0 = c.Waggt + (size_t)j * DEMB;
    const _Float16* wj1 = wj0 + DEMB;
    float y0 = c.bagg[j], y1 = c.bagg[j + 1];
    for (int k = 0; k < DEMB; ++k) {
      const float v = sv[192 + k];
      y0 += v * (float)wj0[k];
      y1 += v * (float)wj1[k];
    }
    y0 = y0 > 0.f ? y0 : 0.f;
    y1 = y1 > 0.f ? y1 : 0.f;
    float* op = c.out + (size_t)g * c.out_stride + c.out_off;
    op[j] = y0; op[j + 1] = y1;
    lds_wave_fence();
  }
}

// Combine: relu(Hcat[512x320] @ c1 + b) -> relu(@ c2 + b) -> out [512x64]
__global__ __launch_bounds__(128) void combine_kernel(
    const float* __restrict__ Hcat, const _Float16* __restrict__ W1t,
    const float* __restrict__ b1, const _Float16* __restrict__ W2t,
    const float* __restrict__ b2, float* __restrict__ out, int nrows) {
  __shared__ __attribute__((aligned(16))) _Float16 s_stage[4][16 * 192];
  const int lane = threadIdx.x & 31;
  const int wave = threadIdx.x >> 5;
  _Float16* stg = s_stage[wave];
  const int tile = blockIdx.x * 4 + wave;
  if (tile * 16 >= nrows) return;
  const int nl = lane & 15, half = (lane >> 4) & 1;
  const float* rowp = Hcat + (size_t)(tile * 16 + nl) * 320;
  v16h a[10];
  for (int kc = 0; kc < 10; ++kc) a[kc] = a_frag_row(rowp, kc, lane);
  v8f h1[12];
  for (int nt = 0; nt < 12; ++nt) {
    v8f acc = acc_bias(b1, nt, lane);
    for (int kc = 0; kc < 10; ++kc)
      acc = wmma_f16(a[kc], b_frag_t(W1t, 320, 0, kc, nt, lane), acc);
    h1[nt] = acc;
  }
  stage_acc(stg, 192, h1, 12, lane, 2);
  lds_wave_fence();
  v16h a2[6];
  for (int kc = 0; kc < 6; ++kc) a2[kc] = a_frag_stage(stg, 192, kc, lane);
  for (int nt = 0; nt < 4; ++nt) {
    v8f acc = acc_bias(b2, nt, lane);
    for (int kc = 0; kc < 6; ++kc)
      acc = wmma_f16(a2[kc], b_frag_t(W2t, 192, 0, kc, nt, lane), acc);
#pragma unroll
    for (int r = 0; r < 8; ++r) {
      float v = acc[r];
      v = v > 0.f ? v : 0.f;
      out[(size_t)(tile * 16 + r + 8 * half) * DEMB + nt * 16 + nl] = v;
    }
  }
}

// ---------------------------------------------------------------------------
// Host launcher. d_in follows setup_inputs() recursive insertion order:
//   params: 0 user_emb, 1 item_emb, 2 rate_emb, 3..16 w1..w7 (W,b),
//     17..20 g_v, 21..24 user_items_att, 25,26 aggre_items,
//     27..30 user_items_att_sf1, 31,32 aggre_items_sf1, 33..36 sf_users_att,
//     37,38 aggre_sf_neigbors, 39..42 user_items_att_s1, 43,44 aggre_items_s1,
//     45..48 user_users_att_s2, 49,50 aggre_neigbors_s2, 51..54 u_user_users_att,
//     55,56 u_aggre_neigbors, 57..60 sf_user_users_att, 61,62 sf_aggre_neigbors,
//     63,64 c1, 65,66 c2
//   67 uids, 68 iids(unused), 69 u_item_pad, 70 u_user_pad, 71 u_user_item_pad,
//   72 sf_user_pad, 73 sf_user_item_pad
// ---------------------------------------------------------------------------
extern "C" void kernel_launch(void* const* d_in, const int* in_sizes, int n_in,
                              void* d_out, int out_size, void* d_ws, size_t ws_size,
                              hipStream_t stream) {
  (void)in_sizes; (void)n_in; (void)out_size; (void)ws_size;
  auto fp = [&](int i) { return (const float*)d_in[i]; };
  auto ip = [&](int i) { return (const int*)d_in[i]; };

  const int B = 512, PI = 50, PU = 30, PUI = 30, PSF = 30, PSFI = 30;

  float* ws = (float*)d_ws;
  float* h_sfI = ws;                                   // [B*PSF, 64]
  float* h_oI  = h_sfI + (size_t)B * PSF * DEMB;       // [B*PU, 64]
  float* Hcat  = h_oI  + (size_t)B * PU * DEMB;        // [B, 320]
  _Float16* wt = (_Float16*)(Hcat + (size_t)B * 320);  // f16 transposed weights

  // ---- weight prep: transposed f16 copies of all WMMA B matrices
  PrepArgs pa{}; unsigned off = 0;
  auto add = [&](int pidx, int din, int dout) -> unsigned {
    pa.e[pa.n].src = fp(pidx); pa.e[pa.n].dstoff = off;
    pa.e[pa.n].din = din; pa.e[pa.n].dout = dout; pa.n++;
    unsigned o = off; off += (unsigned)(din * dout); return o;
  };
  unsigned o_w[7];
  for (int i = 0; i < 7; ++i) o_w[i] = add(3 + 2 * i, 64, 64);
  const unsigned o_gv1 = add(17, 128, 64);
  const unsigned o_gv2 = add(19, 64, 64);
  const int att_idx[7] = {21, 27, 33, 39, 45, 51, 57};
  unsigned o_att[7];
  for (int i = 0; i < 7; ++i) o_att[i] = add(att_idx[i], 128, 64);
  const int agg_idx[7] = {25, 31, 37, 43, 49, 55, 61};
  unsigned o_agg[7];
  for (int i = 0; i < 7; ++i) o_agg[i] = add(agg_idx[i], 64, 64);
  const unsigned o_c1 = add(63, 320, 192);
  const unsigned o_c2 = add(65, 192, 64);
  prep_weights<<<dim3(128), dim3(256), 0, stream>>>(pa, wt);

  BranchCfg base{};
  base.ue = fp(0); base.ie = fp(1); base.re = fp(2);
  base.Wg1t = wt + o_gv1; base.bg1 = fp(18);
  base.Wg2t = wt + o_gv2; base.bg2 = fp(20);

  auto launch = [&](const BranchCfg& cfg) {
    int blocks = (cfg.groups + 7) / 8;
    if (blocks > 1920) blocks = 1920;
    branch_kernel<<<dim3(blocks), dim3(256), 0, stream>>>(cfg);
  };
  // wk index i (w_{i+1}), att index ai, agg index gi
  auto set_w = [&](BranchCfg& c, int wi, int ai, int gi) {
    c.Wkt = wt + o_w[wi];   c.bk   = fp(4 + 2 * wi);
    c.Wa1t = wt + o_att[ai]; c.ba1 = fp(att_idx[ai] + 1);
    c.Wa2 = fp(att_idx[ai] + 2); c.ba2 = fp(att_idx[ai] + 3);
    c.Waggt = wt + o_agg[gi]; c.bagg = fp(agg_idx[gi] + 1);
  };

  { // h_sfI : per (b, friend) item-level aggregation
    BranchCfg c = base;
    c.in_mode = 0; c.pair_ids = ip(73);
    c.mask_ids = ip(73); c.mask_stride = 2;
    c.parent_ids = ip(72); c.parent_stride = 2; c.second_per_row = 0;
    set_w(c, 1, 1, 1);                     // w2, user_items_att_sf1, aggre_items_sf1
    c.out = h_sfI; c.out_stride = DEMB; c.out_off = 0;
    c.groups = B * PSF; c.rows = PSFI;
    launch(c);
  }
  { // h_oI : per (b, social user) item-level aggregation
    BranchCfg c = base;
    c.in_mode = 0; c.pair_ids = ip(71);
    c.mask_ids = ip(71); c.mask_stride = 2;
    c.parent_ids = ip(70); c.parent_stride = 1; c.second_per_row = 0;
    set_w(c, 3, 3, 3);                     // w4, user_items_att_s1, aggre_items_s1
    c.out = h_oI; c.out_stride = DEMB; c.out_off = 0;
    c.groups = B * PU; c.rows = PUI;
    launch(c);
  }
  { // h_iI -> Hcat[:, 0:64]
    BranchCfg c = base;
    c.in_mode = 0; c.pair_ids = ip(69);
    c.mask_ids = ip(69); c.mask_stride = 2;
    c.parent_ids = ip(67); c.parent_stride = 1; c.second_per_row = 0;
    set_w(c, 0, 0, 0);                     // w1, user_items_att, aggre_items
    c.out = Hcat; c.out_stride = 320; c.out_off = 0;
    c.groups = B; c.rows = PI;
    launch(c);
  }
  { // h_iS -> Hcat[:, 64:128]   (per-row second operand)
    BranchCfg c = base;
    c.in_mode = 1; c.in_rows = h_oI;
    c.mask_ids = ip(70); c.mask_stride = 1;
    c.parent_ids = ip(70); c.parent_stride = 1; c.second_per_row = 1;
    set_w(c, 4, 4, 4);                     // w5, user_users_att_s2, aggre_neigbors_s2
    c.out = Hcat; c.out_stride = 320; c.out_off = 64;
    c.groups = B; c.rows = PU;
    launch(c);
  }
  { // h_i_sf -> Hcat[:, 128:192]
    BranchCfg c = base;
    c.in_mode = 1; c.in_rows = h_sfI;
    c.mask_ids = ip(72); c.mask_stride = 2;
    c.parent_ids = ip(67); c.parent_stride = 1; c.second_per_row = 0;
    set_w(c, 2, 2, 2);                     // w3, sf_users_att, aggre_sf_neigbors
    c.out = Hcat; c.out_stride = 320; c.out_off = 128;
    c.groups = B; c.rows = PSF;
    launch(c);
  }
  { // h_su -> Hcat[:, 192:256]
    BranchCfg c = base;
    c.in_mode = 2; c.in_row_ids = ip(70); c.in_row_id_stride = 1;
    c.mask_ids = ip(70); c.mask_stride = 1;
    c.parent_ids = ip(67); c.parent_stride = 1; c.second_per_row = 0;
    set_w(c, 5, 5, 5);                     // w6, u_user_users_att, u_aggre_neigbors
    c.out = Hcat; c.out_stride = 320; c.out_off = 192;
    c.groups = B; c.rows = PU;
    launch(c);
  }
  { // h_sf -> Hcat[:, 256:320]
    BranchCfg c = base;
    c.in_mode = 2; c.in_row_ids = ip(72); c.in_row_id_stride = 2;
    c.mask_ids = ip(72); c.mask_stride = 2;
    c.parent_ids = ip(67); c.parent_stride = 1; c.second_per_row = 0;
    set_w(c, 6, 6, 6);                     // w7, sf_user_users_att, sf_aggre_neigbors
    c.out = Hcat; c.out_stride = 320; c.out_off = 256;
    c.groups = B; c.rows = PSF;
    launch(c);
  }
  // Final combine MLP -> d_out [512, 64] f32
  combine_kernel<<<dim3((B / 16 + 3) / 4), dim3(128), 0, stream>>>(
      Hcat, wt + o_c1, fp(64), wt + o_c2, fp(66), (float*)d_out, B);
}
